// LoraMoeBlock_9474697855506
// MI455X (gfx1250) — compile-verified, matrix operand order checked by reading, not run on hardware
//
#include <hip/hip_runtime.h>
#include <hip/hip_bf16.h>
#include <math.h>

// Problem constants (from reference): B=1, S=1024, H=1024, F=4096, E=8, K=2, R=16
#define Tn    1024
#define Hd    1024
#define Fd    4096
#define Ed    8
#define Rd    16
#define SCAL  2.0f

typedef __attribute__((ext_vector_type(16))) __bf16    bf16x16;
typedef __attribute__((ext_vector_type(8)))  float     v8f;
typedef __attribute__((ext_vector_type(4)))  unsigned  u32x4;

union FragAB { u32x4 u[2]; bf16x16 v; };

static __device__ __forceinline__ unsigned short f2bf(float f) {
  union { float f; unsigned u; } c; c.f = f;
  unsigned r = c.u + 0x7FFFu + ((c.u >> 16) & 1u);   // round-to-nearest-even
  return (unsigned short)(r >> 16);
}
static __device__ __forceinline__ float bf2f(unsigned short b) {
  union { unsigned u; float f; } c; c.u = ((unsigned)b) << 16;
  return c.f;
}
static __device__ __forceinline__ unsigned pk2bf(float lo, float hi) {
  return ((unsigned)f2bf(hi) << 16) | (unsigned)f2bf(lo);
}

// ---------------------------------------------------------------- kernel 1: x f32 -> bf16 (vectorized 4-wide)
__global__ void k_cvt_bf16(const float4* __restrict__ x, uint2* __restrict__ xb, int n4) {
  int i = blockIdx.x * blockDim.x + threadIdx.x;
  if (i < n4) {
    float4 v = x[i];
    uint2 o;
    o.x = pk2bf(v.x, v.y);
    o.y = pk2bf(v.z, v.w);
    xb[i] = o;
  }
}

// ---------------------------------------------------------------- kernel 2: noisy top-k router
// one wave per token; 16 lane-parallel dot products (8 route + 8 noise), shfl reduce,
// softplus-noise, fp32 softmax, top-2, renormalize -> wdense[T][E]; router_logits to d_out tail.
__global__ __launch_bounds__(256)
void k_router(const float* __restrict__ x, const float* __restrict__ noise,
              const float* __restrict__ w_route, const float* __restrict__ w_noise,
              float* __restrict__ rlogits_out, float* __restrict__ wdense) {
  const int wave = threadIdx.x >> 5;
  const int lane = threadIdx.x & 31;
  const int t = blockIdx.x * (blockDim.x >> 5) + wave;
  if (t >= Tn) return;

  float acc[16];
#pragma unroll
  for (int i = 0; i < 16; ++i) acc[i] = 0.f;
  for (int h = lane; h < Hd; h += 32) {
    float xv = x[(size_t)t * Hd + h];
#pragma unroll
    for (int e = 0; e < Ed; ++e) {
      acc[e]     += xv * w_route[(size_t)h * Ed + e];
      acc[8 + e] += xv * w_noise[(size_t)h * Ed + e];
    }
  }
#pragma unroll
  for (int i = 0; i < 16; ++i)
    for (int off = 16; off; off >>= 1)
      acc[i] += __shfl_xor(acc[i], off, 32);

  float rl[Ed], p[Ed];
  float m = -1e30f;
#pragma unroll
  for (int e = 0; e < Ed; ++e) {
    float nl = acc[8 + e];
    float sp = (nl > 20.f) ? nl : log1pf(expf(nl));      // softplus
    rl[e] = acc[e] + noise[(size_t)t * Ed + e] * sp;
    m = fmaxf(m, rl[e]);
  }
  float s = 0.f;
#pragma unroll
  for (int e = 0; e < Ed; ++e) { p[e] = expf(rl[e] - m); s += p[e]; }
  float inv = 1.f / s;
#pragma unroll
  for (int e = 0; e < Ed; ++e) p[e] *= inv;
  int i0 = 0;
#pragma unroll
  for (int e = 1; e < Ed; ++e) if (p[e] > p[i0]) i0 = e;
  int i1 = (i0 == 0) ? 1 : 0;
#pragma unroll
  for (int e = 0; e < Ed; ++e) if (e != i0 && p[e] > p[i1]) i1 = e;
  float s2 = p[i0] + p[i1];

  if (lane < Ed) {
    rlogits_out[(size_t)t * Ed + lane] = rl[lane];
    float w = (lane == i0) ? p[i0] / s2 : ((lane == i1) ? p[i1] / s2 : 0.f);
    wdense[(size_t)t * Ed + lane] = w;
  }
}

// ---------------------------------------------------------------- kernel 3: xa = 2 * (x @ A[e])  -> bf16 [E][T][R]
// one thread per (e, t, group of 4 ranks): float4 loads of A rows, packed bf16 store.
__global__ void k_xa(const float* __restrict__ x, const float* __restrict__ a,
                     unsigned short* __restrict__ xa) {
  int idx = blockIdx.x * blockDim.x + threadIdx.x;     // over E*T*(R/4)
  if (idx >= Ed * Tn * (Rd / 4)) return;
  int rq = idx % (Rd / 4);
  int t  = (idx / (Rd / 4)) % Tn;
  int e  = idx / ((Rd / 4) * Tn);
  const float* xp = x + (size_t)t * Hd;
  const float* ap = a + ((size_t)e * Hd) * Rd + rq * 4;
  float a0 = 0.f, a1 = 0.f, a2 = 0.f, a3 = 0.f;
  for (int h = 0; h < Hd; ++h) {
    float xv = xp[h];
    float4 av = *reinterpret_cast<const float4*>(ap + (size_t)h * Rd);
    a0 += xv * av.x; a1 += xv * av.y; a2 += xv * av.z; a3 += xv * av.w;
  }
  uint2 o;
  o.x = pk2bf(SCAL * a0, SCAL * a1);
  o.y = pk2bf(SCAL * a2, SCAL * a3);
  *reinterpret_cast<uint2*>(xa + ((size_t)e * Tn + t) * Rd + rq * 4) = o;
}

// ---------------------------------------------------------------- kernel 5b: xad = 2 * (act[e] @ a_down[e]) -> bf16 [E][T][R]
__global__ void k_xa_down(const unsigned short* __restrict__ act, const float* __restrict__ a_down,
                          unsigned short* __restrict__ xad) {
  int idx = blockIdx.x * blockDim.x + threadIdx.x;     // over E*T*(R/4)
  if (idx >= Ed * Tn * (Rd / 4)) return;
  int rq = idx % (Rd / 4);
  int t  = (idx / (Rd / 4)) % Tn;
  int e  = idx / ((Rd / 4) * Tn);
  const unsigned short* ac = act + ((size_t)e * Tn + t) * Fd;
  const float* ap = a_down + ((size_t)e * Fd) * Rd + rq * 4;
  float a0 = 0.f, a1 = 0.f, a2 = 0.f, a3 = 0.f;
  for (int f = 0; f < Fd; ++f) {
    float av = bf2f(ac[f]);
    float4 dv = *reinterpret_cast<const float4*>(ap + (size_t)f * Rd);
    a0 += av * dv.x; a1 += av * dv.y; a2 += av * dv.z; a3 += av * dv.w;
  }
  uint2 o;
  o.x = pk2bf(SCAL * a0, SCAL * a1);
  o.y = pk2bf(SCAL * a2, SCAL * a3);
  *reinterpret_cast<uint2*>(xad + ((size_t)e * Tn + t) * Rd + rq * 4) = o;
}

// ---------------------------------------------------------------- kernel 4: act[e] = silu(x@Wg + 2*xa_g@Bg) * (x@Wu + 2*xa_u@Bu)
// 128(T)x64(F) tile per block; 8 waves in 4x2; each wave 2x2 WMMA 16x16 tiles, dual gate/up acc.
// LoRA is one extra K=32 WMMA step: A = [2*xa_g | 2*xa_u], Bg = [b_gate | 0], Bu = [0 | b_up].
#define CT 128
#define CF 64
__global__ __launch_bounds__(256)
void k_act(const unsigned short* __restrict__ xb,
           const float* __restrict__ w_gate, const float* __restrict__ w_up,
           const unsigned short* __restrict__ xag, const unsigned short* __restrict__ xau,
           const float* __restrict__ b_gate, const float* __restrict__ b_up,
           unsigned short* __restrict__ act) {
  __shared__ __align__(16) unsigned short xs[CT * 32];   // [t][k] bf16
  __shared__ __align__(16) unsigned short wg[CF * 32];   // [f][k] bf16 (K-contiguous)
  __shared__ __align__(16) unsigned short wu[CF * 32];

  const int tid  = threadIdx.x;
  const int lane = tid & 31, wid = tid >> 5;
  const int lid  = lane & 15, hi = lane >> 4;
  const int waveM = wid >> 1, waveN = wid & 1;
  const int e = blockIdx.z;
  const int tBase = blockIdx.y * CT;
  const int fBase = blockIdx.x * CF;

  v8f accG[2][2], accU[2][2];
#pragma unroll
  for (int sm = 0; sm < 2; ++sm)
#pragma unroll
    for (int sn = 0; sn < 2; ++sn)
#pragma unroll
      for (int i = 0; i < 8; ++i) { accG[sm][sn][i] = 0.f; accU[sm][sn][i] = 0.f; }

  for (int kb = 0; kb < Hd; kb += 32) {
#pragma unroll
    for (int i = 0; i < 2; ++i) {                      // x tile 128x32 bf16, b128 loads
      int chunk = tid + i * 256;
      int r = chunk >> 2, c = (chunk & 3) * 8;
      *reinterpret_cast<u32x4*>(&xs[r * 32 + c]) =
          *reinterpret_cast<const u32x4*>(&xb[(size_t)(tBase + r) * Hd + kb + c]);
    }
#pragma unroll
    for (int i = 0; i < 2; ++i) {                      // weight tiles 32x64: float4 loads, bf16 transpose store
      int ch = tid + i * 256;                          // 0..511 over 512 float4 chunks
      int k = ch >> 4, f4 = (ch & 15) * 4;
      float4 g = *reinterpret_cast<const float4*>(&w_gate[(size_t)(kb + k) * Fd + fBase + f4]);
      float4 u = *reinterpret_cast<const float4*>(&w_up  [(size_t)(kb + k) * Fd + fBase + f4]);
      wg[(f4 + 0) * 32 + k] = f2bf(g.x);
      wg[(f4 + 1) * 32 + k] = f2bf(g.y);
      wg[(f4 + 2) * 32 + k] = f2bf(g.z);
      wg[(f4 + 3) * 32 + k] = f2bf(g.w);
      wu[(f4 + 0) * 32 + k] = f2bf(u.x);
      wu[(f4 + 1) * 32 + k] = f2bf(u.y);
      wu[(f4 + 2) * 32 + k] = f2bf(u.z);
      wu[(f4 + 3) * 32 + k] = f2bf(u.w);
    }
    if (kb + 32 < Hd) {
      __builtin_prefetch(&w_gate[(size_t)(kb + 32) * Fd + fBase], 0, 1);
      __builtin_prefetch(&w_up  [(size_t)(kb + 32) * Fd + fBase], 0, 1);
    }
    __syncthreads();

    FragAB a[2], bg[2], bu[2];
#pragma unroll
    for (int sm = 0; sm < 2; ++sm) {
      int r = waveM * 32 + sm * 16 + lid;
      a[sm].u[0] = *reinterpret_cast<const u32x4*>(&xs[r * 32 + hi * 8]);
      a[sm].u[1] = *reinterpret_cast<const u32x4*>(&xs[r * 32 + 16 + hi * 8]);
    }
#pragma unroll
    for (int sn = 0; sn < 2; ++sn) {
      int f = waveN * 32 + sn * 16 + lid;
      bg[sn].u[0] = *reinterpret_cast<const u32x4*>(&wg[f * 32 + hi * 8]);
      bg[sn].u[1] = *reinterpret_cast<const u32x4*>(&wg[f * 32 + 16 + hi * 8]);
      bu[sn].u[0] = *reinterpret_cast<const u32x4*>(&wu[f * 32 + hi * 8]);
      bu[sn].u[1] = *reinterpret_cast<const u32x4*>(&wu[f * 32 + 16 + hi * 8]);
    }
#pragma unroll
    for (int sm = 0; sm < 2; ++sm)
#pragma unroll
      for (int sn = 0; sn < 2; ++sn) {
        accG[sm][sn] = __builtin_amdgcn_wmma_f32_16x16x32_bf16(
            false, a[sm].v, false, bg[sn].v, (short)0, accG[sm][sn], false, false);
        accU[sm][sn] = __builtin_amdgcn_wmma_f32_16x16x32_bf16(
            false, a[sm].v, false, bu[sn].v, (short)0, accU[sm][sn], false, false);
      }
    __syncthreads();
  }

  // ---- LoRA K-extension (one K=32 step): A = [2*xa_g | 2*xa_u], Bg = [b_gate | 0], Bu = [0 | b_up]
#pragma unroll
  for (int i = 0; i < 2; ++i) {
    int chunk = tid + i * 256;
    int r = chunk >> 2, part = chunk & 3;              // 0,1 -> xa_g halves ; 2,3 -> xa_u halves
    const unsigned short* src = (part < 2) ? xag : xau;
    int rr = (part & 1) * 8;
    int dst = (part < 2) ? rr : 16 + rr;
    *reinterpret_cast<u32x4*>(&xs[r * 32 + dst]) =
        *reinterpret_cast<const u32x4*>(&src[((size_t)e * Tn + tBase + r) * Rd + rr]);
  }
  {
    int k = tid >> 4, f4 = (tid & 15) * 4;             // 256 float4 chunks cover 16x64
    float4 g = *reinterpret_cast<const float4*>(&b_gate[((size_t)e * Rd + k) * Fd + fBase + f4]);
    float4 u = *reinterpret_cast<const float4*>(&b_up  [((size_t)e * Rd + k) * Fd + fBase + f4]);
#pragma unroll
    for (int j = 0; j < 4; ++j) {
      wg[(f4 + j) * 32 + 16 + k] = 0;                  // upper half of Bg is zero
      wu[(f4 + j) * 32 + k]      = 0;                  // lower half of Bu is zero
    }
    wg[(f4 + 0) * 32 + k] = f2bf(g.x);
    wg[(f4 + 1) * 32 + k] = f2bf(g.y);
    wg[(f4 + 2) * 32 + k] = f2bf(g.z);
    wg[(f4 + 3) * 32 + k] = f2bf(g.w);
    wu[(f4 + 0) * 32 + 16 + k] = f2bf(u.x);
    wu[(f4 + 1) * 32 + 16 + k] = f2bf(u.y);
    wu[(f4 + 2) * 32 + 16 + k] = f2bf(u.z);
    wu[(f4 + 3) * 32 + 16 + k] = f2bf(u.w);
  }
  __syncthreads();
  {
    FragAB a[2], bg[2], bu[2];
#pragma unroll
    for (int sm = 0; sm < 2; ++sm) {
      int r = waveM * 32 + sm * 16 + lid;
      a[sm].u[0] = *reinterpret_cast<const u32x4*>(&xs[r * 32 + hi * 8]);
      a[sm].u[1] = *reinterpret_cast<const u32x4*>(&xs[r * 32 + 16 + hi * 8]);
    }
#pragma unroll
    for (int sn = 0; sn < 2; ++sn) {
      int f = waveN * 32 + sn * 16 + lid;
      bg[sn].u[0] = *reinterpret_cast<const u32x4*>(&wg[f * 32 + hi * 8]);
      bg[sn].u[1] = *reinterpret_cast<const u32x4*>(&wg[f * 32 + 16 + hi * 8]);
      bu[sn].u[0] = *reinterpret_cast<const u32x4*>(&wu[f * 32 + hi * 8]);
      bu[sn].u[1] = *reinterpret_cast<const u32x4*>(&wu[f * 32 + 16 + hi * 8]);
    }
#pragma unroll
    for (int sm = 0; sm < 2; ++sm)
#pragma unroll
      for (int sn = 0; sn < 2; ++sn) {
        accG[sm][sn] = __builtin_amdgcn_wmma_f32_16x16x32_bf16(
            false, a[sm].v, false, bg[sn].v, (short)0, accG[sm][sn], false, false);
        accU[sm][sn] = __builtin_amdgcn_wmma_f32_16x16x32_bf16(
            false, a[sm].v, false, bu[sn].v, (short)0, accU[sm][sn], false, false);
      }
  }
  __syncthreads();

  // epilogue: act = silu(gate) * up -> bf16.  C layout: m = hi*8 + v, n = lid.
#pragma unroll
  for (int sm = 0; sm < 2; ++sm)
#pragma unroll
    for (int sn = 0; sn < 2; ++sn)
#pragma unroll
      for (int v = 0; v < 8; ++v) {
        float g = accG[sm][sn][v];
        float u = accU[sm][sn][v];
        float s = g / (1.f + expf(-g));
        int t = tBase + waveM * 32 + sm * 16 + hi * 8 + v;
        int f = fBase + waveN * 32 + sn * 16 + lid;
        act[((size_t)e * Tn + t) * Fd + f] = f2bf(s * u);
      }
}

// ---------------------------------------------------------------- kernel 6: down GEMM + LoRA + weighted combine
// 64(T)x64(H) tile; 8 waves 4x2 (wave = 16 rows x 32 cols = 1x2 WMMA). Experts looped inside,
// each expert's partial scaled by wdense[t,e] and accumulated in registers. f32 store.
#define DT 64
#define DH 64
__global__ __launch_bounds__(256)
void k_down(const unsigned short* __restrict__ act, const float* __restrict__ w_down,
            const unsigned short* __restrict__ xad, const float* __restrict__ b_down,
            const float* __restrict__ wdense, float* __restrict__ out) {
  __shared__ __align__(16) unsigned short as_[DT * 32];  // [t][k]
  __shared__ __align__(16) unsigned short wd[DH * 32];   // [h][k]
  __shared__ float wde[DT * Ed];

  const int tid  = threadIdx.x;
  const int lane = tid & 31, wid = tid >> 5;
  const int lid  = lane & 15, hi = lane >> 4;
  const int waveM = wid >> 1, waveN = wid & 1;
  const int tBase = blockIdx.y * DT;
  const int hBase = blockIdx.x * DH;

  for (int i = tid; i < DT * Ed; i += 256)
    wde[i] = wdense[(size_t)(tBase + (i >> 3)) * Ed + (i & 7)];

  v8f oacc[2];
#pragma unroll
  for (int sn = 0; sn < 2; ++sn)
#pragma unroll
    for (int i = 0; i < 8; ++i) oacc[sn][i] = 0.f;

  for (int e = 0; e < Ed; ++e) {
    v8f eacc[2];
#pragma unroll
    for (int sn = 0; sn < 2; ++sn)
#pragma unroll
      for (int i = 0; i < 8; ++i) eacc[sn][i] = 0.f;

    for (int kb = 0; kb < Fd; kb += 32) {
      {                                                 // act tile 64x32 bf16 (already bf16), b128 loads
        int r = tid >> 2, c = (tid & 3) * 8;
        *reinterpret_cast<u32x4*>(&as_[r * 32 + c]) =
            *reinterpret_cast<const u32x4*>(&act[((size_t)e * Tn + tBase + r) * Fd + kb + c]);
      }
#pragma unroll
      for (int i = 0; i < 2; ++i) {                     // w_down 32x64: float4 loads, bf16 transpose store
        int ch = tid + i * 256;
        int k = ch >> 4, h4 = (ch & 15) * 4;
        float4 wv = *reinterpret_cast<const float4*>(&w_down[(size_t)(kb + k) * Hd + hBase + h4]);
        wd[(h4 + 0) * 32 + k] = f2bf(wv.x);
        wd[(h4 + 1) * 32 + k] = f2bf(wv.y);
        wd[(h4 + 2) * 32 + k] = f2bf(wv.z);
        wd[(h4 + 3) * 32 + k] = f2bf(wv.w);
      }
      if (kb + 32 < Fd)
        __builtin_prefetch(&w_down[(size_t)(kb + 32) * Hd + hBase], 0, 1);
      __syncthreads();

      FragAB a, b[2];
      int r = waveM * 16 + lid;
      a.u[0] = *reinterpret_cast<const u32x4*>(&as_[r * 32 + hi * 8]);
      a.u[1] = *reinterpret_cast<const u32x4*>(&as_[r * 32 + 16 + hi * 8]);
#pragma unroll
      for (int sn = 0; sn < 2; ++sn) {
        int h = waveN * 32 + sn * 16 + lid;
        b[sn].u[0] = *reinterpret_cast<const u32x4*>(&wd[h * 32 + hi * 8]);
        b[sn].u[1] = *reinterpret_cast<const u32x4*>(&wd[h * 32 + 16 + hi * 8]);
        eacc[sn] = __builtin_amdgcn_wmma_f32_16x16x32_bf16(
            false, a.v, false, b[sn].v, (short)0, eacc[sn], false, false);
      }
      __syncthreads();
    }

    // ---- LoRA down extension (one K=32 step): A = [2*xad | 0], B = [b_down | 0] ----
    {
      int r = tid >> 2, c = (tid & 3) * 8;
      if (c < Rd) {
        *reinterpret_cast<u32x4*>(&as_[r * 32 + c]) =
            *reinterpret_cast<const u32x4*>(&xad[((size_t)e * Tn + tBase + r) * Rd + c]);
      } else {
        u32x4 z = {0u, 0u, 0u, 0u};
        *reinterpret_cast<u32x4*>(&as_[r * 32 + c]) = z;
      }
    }
    {
      int k = tid >> 4, h4 = (tid & 15) * 4;           // 256 float4 chunks cover 16x64
      float4 bv = *reinterpret_cast<const float4*>(&b_down[((size_t)e * Rd + k) * Hd + hBase + h4]);
#pragma unroll
      for (int j = 0; j < 4; ++j) wd[(h4 + j) * 32 + 16 + k] = 0;
      wd[(h4 + 0) * 32 + k] = f2bf(bv.x);
      wd[(h4 + 1) * 32 + k] = f2bf(bv.y);
      wd[(h4 + 2) * 32 + k] = f2bf(bv.z);
      wd[(h4 + 3) * 32 + k] = f2bf(bv.w);
    }
    __syncthreads();
    {
      FragAB a, b[2];
      int r = waveM * 16 + lid;
      a.u[0] = *reinterpret_cast<const u32x4*>(&as_[r * 32 + hi * 8]);
      a.u[1] = *reinterpret_cast<const u32x4*>(&as_[r * 32 + 16 + hi * 8]);
#pragma unroll
      for (int sn = 0; sn < 2; ++sn) {
        int h = waveN * 32 + sn * 16 + lid;
        b[sn].u[0] = *reinterpret_cast<const u32x4*>(&wd[h * 32 + hi * 8]);
        b[sn].u[1] = *reinterpret_cast<const u32x4*>(&wd[h * 32 + 16 + hi * 8]);
        eacc[sn] = __builtin_amdgcn_wmma_f32_16x16x32_bf16(
            false, a.v, false, b[sn].v, (short)0, eacc[sn], false, false);
      }
    }
    __syncthreads();

    // scale by router weight and accumulate
#pragma unroll
    for (int sn = 0; sn < 2; ++sn)
#pragma unroll
      for (int v = 0; v < 8; ++v) {
        int mloc = waveM * 16 + hi * 8 + v;
        oacc[sn][v] += wde[mloc * Ed + e] * eacc[sn][v];
      }
  }

#pragma unroll
  for (int sn = 0; sn < 2; ++sn)
#pragma unroll
    for (int v = 0; v < 8; ++v) {
      int t = tBase + waveM * 16 + hi * 8 + v;
      int h = hBase + waveN * 32 + sn * 16 + lid;
      out[(size_t)t * Hd + h] = oacc[sn][v];
    }
}

// ---------------------------------------------------------------- host launcher
extern "C" void kernel_launch(void* const* d_in, const int* in_sizes, int n_in,
                              void* d_out, int out_size, void* d_ws, size_t ws_size,
                              hipStream_t stream) {
  (void)in_sizes; (void)n_in; (void)out_size; (void)ws_size;
  const float* x        = (const float*)d_in[0];
  const float* noise    = (const float*)d_in[1];
  const float* w_route  = (const float*)d_in[2];
  const float* w_noise  = (const float*)d_in[3];
  const float* w_gate   = (const float*)d_in[4];
  const float* w_up     = (const float*)d_in[5];
  const float* w_down   = (const float*)d_in[6];
  const float* a_gate   = (const float*)d_in[7];
  const float* b_gate   = (const float*)d_in[8];
  const float* a_up     = (const float*)d_in[9];
  const float* b_up     = (const float*)d_in[10];
  const float* a_down   = (const float*)d_in[11];
  const float* b_down   = (const float*)d_in[12];

  float* out_main = (float*)d_out;                     // [T][H]
  float* out_rl   = out_main + (size_t)Tn * Hd;        // [T][E]

  // workspace carve-out (256B aligned)
  char* ws = (char*)d_ws;
  size_t off = 0;
  auto carve = [&](size_t bytes) -> char* {
    char* p = ws + off;
    off += (bytes + 255) & ~(size_t)255;
    return p;
  };
  unsigned short* xb   = (unsigned short*)carve((size_t)Tn * Hd * 2);        // x in bf16
  float*          wden = (float*)carve((size_t)Tn * Ed * 4);                 // dense router weights
  unsigned short* xag  = (unsigned short*)carve((size_t)Ed * Tn * Rd * 2);   // 2*x@a_gate bf16
  unsigned short* xau  = (unsigned short*)carve((size_t)Ed * Tn * Rd * 2);
  unsigned short* xad  = (unsigned short*)carve((size_t)Ed * Tn * Rd * 2);   // 2*act@a_down bf16
  unsigned short* actb = (unsigned short*)carve((size_t)Ed * Tn * Fd * 2);   // act bf16 (64 MB)

  // 1) x -> bf16 (4-wide)
  k_cvt_bf16<<<(Tn * Hd / 4 + 255) / 256, 256, 0, stream>>>(
      (const float4*)x, (uint2*)xb, Tn * Hd / 4);
  // 2) router -> wdense + router_logits
  k_router<<<Tn / 8, 256, 0, stream>>>(x, noise, w_route, w_noise, out_rl, wden);
  // 3) LoRA-A projections for gate/up (4 ranks per thread)
  k_xa<<<(Ed * Tn * (Rd / 4)) / 256, 256, 0, stream>>>(x, a_gate, xag);
  k_xa<<<(Ed * Tn * (Rd / 4)) / 256, 256, 0, stream>>>(x, a_up, xau);
  // 4) fused gate/up GEMM + LoRA + SiLU*up -> act
  dim3 gAct(Fd / CF, Tn / CT, Ed);
  k_act<<<gAct, 256, 0, stream>>>(xb, w_gate, w_up, xag, xau, b_gate, b_up, actb);
  // 5) LoRA-A projection for down
  k_xa_down<<<(Ed * Tn * (Rd / 4)) / 256, 256, 0, stream>>>(actb, a_down, xad);
  // 6) down GEMM + LoRA + router-weighted combine -> out
  dim3 gDown(Hd / DH, Tn / DT);
  k_down<<<gDown, 256, 0, stream>>>(actb, w_down, xad, b_down, wden, out_main);
}
